// DeformableAtt_68521908240977
// MI455X (gfx1250) — matrix operand order repaired and checked by reading, not compiled, not run on hardware
//
#include <hip/hip_runtime.h>
#include <hip/hip_bf16.h>

// ---------------- problem constants ----------------
#define BB    4
#define DIMC  128
#define HH    56
#define WW    56
#define HW    3136      // H*W == n (stride-1 'same' conv)
#define NHD   4
#define HCC   32

typedef __attribute__((ext_vector_type(16))) __bf16 v16bf;
typedef __attribute__((ext_vector_type(8)))  float  v8f;

union Frag16 {
  v16bf v;
  uint4 q[2];
  unsigned short u[16];
};

__device__ __forceinline__ unsigned short f2bf(float f) {
  unsigned int u = __float_as_uint(f);
  u = u + 0x7FFFu + ((u >> 16) & 1u);   // round-to-nearest-even
  return (unsigned short)(u >> 16);
}

// ---------------- prep: weights fp32 -> bf16 ----------------
__global__ void cvt_w_kernel(const float* __restrict__ qw, const float* __restrict__ kw,
                             const float* __restrict__ vw, const float* __restrict__ ow,
                             unsigned short* __restrict__ dst) {
  int t = blockIdx.x * blockDim.x + threadIdx.x;
  if (t >= 4 * 16384) return;
  int which = t >> 14, idx = t & 16383;
  const float* s = (which == 0) ? qw : (which == 1) ? kw : (which == 2) ? vw : ow;
  dst[t] = f2bf(s[idx]);
}

// ---------------- prep: x [B][128][HW] fp32 -> xT [B][HW][128] bf16 ----------------
__global__ void xpose_kernel(const float* __restrict__ x, unsigned short* __restrict__ xT) {
  int t = blockIdx.x * blockDim.x + threadIdx.x;
  if (t >= BB * HW * 128) return;
  int c = t & 127;
  int rem = t >> 7;
  int m = rem % HW;
  int b = rem / HW;
  xT[t] = f2bf(x[((size_t)b * 128 + c) * HW + m]);
}

// ---------------- generic 128xN bf16 WMMA GEMM ----------------
// D = A(128x128) * B(128xN) + bias.  B supplied transposed: BT [batch][N][128].
// Epilogues: fp32 [batch][128][N], bf16 [batch][128][N], bf16-transposed [batch][N][128].
// All 8 A-fragment loads of a K-step are issued before the WMMA chain so the
// matrix pipe runs while later fragments are still in flight.
__global__ void __launch_bounds__(128) gemm128_bf16(
    const unsigned short* __restrict__ A,   // [128][128] bf16  (rows=M, cols=K)
    const unsigned short* __restrict__ BT,  // [batch][N][128]  (rows=N, cols=K)
    const float* __restrict__ bias,         // [128]
    int N,
    float* __restrict__ outF,
    unsigned short* __restrict__ outBF,
    unsigned short* __restrict__ outT) {
  const int lane = threadIdx.x & 31;
  const int wave = threadIdx.x >> 5;
  const int batch = blockIdx.y;
  const int nt = blockIdx.x * 4 + wave;
  const int n0 = nt * 16;
  if (n0 >= N) return;
  const int l15 = lane & 15;
  const bool lo = lane < 16;

  const unsigned short* bt = BT + (size_t)batch * N * 128;

  v8f acc[8];
#pragma unroll
  for (int i = 0; i < 8; ++i) acc[i] = (v8f)(0.f);

#pragma unroll
  for (int kk = 0; kk < 4; ++kk) {
    const int k0 = kk * 32;
    // B fragment 32x16: lane<16 -> col n0+l, K=k0..k0+15; lane>=16 -> col n0+l-16, K=k0+16..31
    Frag16 bf;
    const unsigned short* bp = bt + (size_t)(n0 + l15) * 128 + k0 + (lo ? 0 : 16);
    bf.q[0] = *(const uint4*)(bp);
    bf.q[1] = *(const uint4*)(bp + 8);
    // stage all 8 A fragments first (loads batched -> overlapped with WMMAs)
    Frag16 af[8];
#pragma unroll
    for (int mt = 0; mt < 8; ++mt) {
      const unsigned short* ap = A + (size_t)(mt * 16 + l15) * 128 + k0 + (lo ? 0 : 16);
      af[mt].q[0] = *(const uint4*)(ap);
      af[mt].q[1] = *(const uint4*)(ap + 8);
    }
#pragma unroll
    for (int mt = 0; mt < 8; ++mt) {
      acc[mt] = __builtin_amdgcn_wmma_f32_16x16x32_bf16(false, af[mt].v, false, bf.v,
                                                        (short)0, acc[mt], false, false);
    }
  }

  const int off8 = lo ? 0 : 8;
#pragma unroll
  for (int mt = 0; mt < 8; ++mt) {
    const int mbase = mt * 16 + off8;
    float4 b0 = *(const float4*)(bias + mbase);
    float4 b1 = *(const float4*)(bias + mbase + 4);
    float bv[8] = {b0.x, b0.y, b0.z, b0.w, b1.x, b1.y, b1.z, b1.w};
    float r[8];
#pragma unroll
    for (int i = 0; i < 8; ++i) r[i] = acc[mt][i] + bv[i];
    if (outF) {
      float* of = outF + (size_t)batch * 128 * N;
#pragma unroll
      for (int i = 0; i < 8; ++i) of[(size_t)(mbase + i) * N + n0 + l15] = r[i];
    }
    if (outBF) {
      unsigned short* ob = outBF + (size_t)batch * 128 * N;
#pragma unroll
      for (int i = 0; i < 8; ++i) ob[(size_t)(mbase + i) * N + n0 + l15] = f2bf(r[i]);
    }
    if (outT) {
      unsigned short* ot = outT + (size_t)batch * N * 128 + (size_t)(n0 + l15) * 128 + mbase;
      Frag16 pk;
#pragma unroll
      for (int i = 0; i < 8; ++i) pk.u[i] = f2bf(r[i]);
      *(uint4*)ot = pk.q[0];
    }
  }
}

// ---------------- offset branch: depthwise 7x7 conv ----------------
// q viewed as [8][64][HW]; output channel-last hn [8*HW][64]
__global__ void __launch_bounds__(64) dwconv7_kernel(
    const float* __restrict__ q, const float* __restrict__ dw_w,
    const float* __restrict__ dw_b, float* __restrict__ hn) {
  const int c = threadIdx.x;
  const int pj = blockIdx.x;           // bg*HW + j
  const int j = pj % HW;
  const int bg = pj / HW;
  const int y = j / WW, x = j % WW;
  const float* plane = q + ((size_t)bg * 64 + c) * HW;
  const float* w = dw_w + c * 49;
  float s = dw_b[c];
#pragma unroll
  for (int dy = 0; dy < 7; ++dy) {
    int yy = y + dy - 3;
    if (yy < 0 || yy >= HH) continue;
#pragma unroll
    for (int dx = 0; dx < 7; ++dx) {
      int xx = x + dx - 3;
      if (xx < 0 || xx >= WW) continue;
      s += plane[yy * WW + xx] * w[dy * 7 + dx];
    }
  }
  hn[(size_t)pj * 64 + c] = s;
}

// ---------------- offset branch: LN + GELU + 1x1 -> tanh -> sample coords ----------------
__global__ void __launch_bounds__(64) lnoff_kernel(
    const float* __restrict__ hn, const float* __restrict__ ln_g,
    const float* __restrict__ ln_b, const float* __restrict__ off_w,
    float* __restrict__ pos) {     // [8*HW][2] = (gx_pix, gy_pix)
  __shared__ float red[64];
  const int c = threadIdx.x;
  const int pj = blockIdx.x;
  float v = hn[(size_t)pj * 64 + c];
  red[c] = v; __syncthreads();
  for (int s = 32; s > 0; s >>= 1) { if (c < s) red[c] += red[c + s]; __syncthreads(); }
  float mu = red[0] * (1.0f / 64.0f); __syncthreads();
  float d = v - mu;
  red[c] = d * d; __syncthreads();
  for (int s = 32; s > 0; s >>= 1) { if (c < s) red[c] += red[c + s]; __syncthreads(); }
  float var = red[0] * (1.0f / 64.0f); __syncthreads();
  float hb = d * rsqrtf(var + 1e-5f) * ln_g[c] + ln_b[c];
  hb = 0.5f * hb * (1.0f + erff(hb * 0.70710678118f));   // exact GELU
  red[c] = off_w[c] * hb; __syncthreads();
  for (int s = 32; s > 0; s >>= 1) { if (c < s) red[c] += red[c + s]; __syncthreads(); }
  float o0 = red[0]; __syncthreads();
  red[c] = off_w[64 + c] * hb; __syncthreads();
  for (int s = 32; s > 0; s >>= 1) { if (c < s) red[c] += red[c + s]; __syncthreads(); }
  if (c == 0) {
    float o1 = red[0];
    const int j = pj % HW;
    const int y = j / WW, x = j % WW;
    float offy = tanhf(o0) * (2.0f / 56.0f);   // tanh * (1/Hk) * ORF
    float offx = tanhf(o1) * (2.0f / 56.0f);
    float ry = ((float)y + 0.5f) / 56.0f * 2.0f - 1.0f;
    float rx = ((float)x + 0.5f) / 56.0f * 2.0f - 1.0f;
    float gy = offy + ry, gx = offx + rx;           // normalized (y,x)
    pos[(size_t)pj * 2 + 0] = (gx + 1.0f) * 0.5f * 55.0f;  // align_corners=True pixel coords
    pos[(size_t)pj * 2 + 1] = (gy + 1.0f) * 0.5f * 55.0f;
  }
}

// ---------------- deformable bilinear gather -> xsT [B][HW][128] bf16 ----------------
__global__ void gather_kernel(const float* __restrict__ x, const float* __restrict__ pos,
                              unsigned short* __restrict__ xsT) {
  int t = blockIdx.x * blockDim.x + threadIdx.x;
  if (t >= 8 * HW) return;
  int bg = t / HW, j = t % HW;
  int b = bg >> 1, g = bg & 1;
  float gx = pos[(size_t)t * 2], gy = pos[(size_t)t * 2 + 1];
  float x0f = floorf(gx), y0f = floorf(gy);
  float wx1 = gx - x0f, wx0 = 1.f - wx1;
  float wy1 = gy - y0f, wy0 = 1.f - wy1;
  int x0 = (int)x0f, y0 = (int)y0f, x1 = x0 + 1, y1 = y0 + 1;
  float vX0 = (x0 >= 0 && x0 <= 55) ? 1.f : 0.f;
  float vX1 = (x1 >= 0 && x1 <= 55) ? 1.f : 0.f;
  float vY0 = (y0 >= 0 && y0 <= 55) ? 1.f : 0.f;
  float vY1 = (y1 >= 0 && y1 <= 55) ? 1.f : 0.f;
  int xc0 = min(max(x0, 0), 55), xc1 = min(max(x1, 0), 55);
  int yc0 = min(max(y0, 0), 55), yc1 = min(max(y1, 0), 55);
  int i00 = yc0 * WW + xc0, i01 = yc0 * WW + xc1;
  int i10 = yc1 * WW + xc0, i11 = yc1 * WW + xc1;
  float w00 = wx0 * wy0 * vX0 * vY0, w01 = wx1 * wy0 * vX1 * vY0;
  float w10 = wx0 * wy1 * vX0 * vY1, w11 = wx1 * wy1 * vX1 * vY1;
  const float* base = x + ((size_t)b * 128 + g * 64) * HW;
  unsigned short* dst = xsT + ((size_t)b * HW + j) * 128 + g * 64;
#pragma unroll 4
  for (int c = 0; c < 64; ++c) {
    const float* p = base + (size_t)c * HW;
    float val = p[i00] * w00 + p[i01] * w01 + p[i10] * w10 + p[i11] * w11;
    dst[c] = f2bf(val);
  }
}

// ---------------- flash attention (bf16 WMMA, online softmax) ----------------
// qT [B][HW][128], kT [B][HW][128], v [B][128][HW] -> outT [B][HW][128]
__global__ void __launch_bounds__(128) attn_kernel(
    const unsigned short* __restrict__ qT, const unsigned short* __restrict__ kT,
    const unsigned short* __restrict__ vB, unsigned short* __restrict__ outT) {
  __shared__ __align__(16) unsigned short pbuf[4][512];   // per-wave 16x32 bf16 P tile
  const int lane = threadIdx.x & 31;
  const int wave = threadIdx.x >> 5;
  const int t = blockIdx.x * 4 + wave;        // 0..3135
  const int b = t / (NHD * 196);
  const int r1 = t % (NHD * 196);
  const int h = r1 / 196;
  const int mt = r1 % 196;
  const int m0 = mt * 16;
  const int l15 = lane & 15;
  const bool lo = lane < 16;
  const float scale = 0.17677669529663687f;   // 32^-0.5

  // Q fragment (16x32, whole head dim) held for entire loop
  Frag16 aq;
  {
    const unsigned short* qp =
        qT + ((size_t)(b * HW + m0 + l15)) * 128 + h * 32 + (lo ? 0 : 16);
    aq.q[0] = *(const uint4*)qp;
    aq.q[1] = *(const uint4*)(qp + 8);
  }

  v8f acc0 = (v8f)(0.f), acc1 = (v8f)(0.f);
  float mrow[8], lrow[8];
#pragma unroll
  for (int i = 0; i < 8; ++i) { mrow[i] = -3.0e38f; lrow[i] = 0.f; }

  const unsigned short* kbase = kT + (size_t)b * HW * 128 + h * 32 + (lo ? 0 : 16);
  const unsigned short* vbase = vB + ((size_t)b * 128 + h * 32) * HW + (lo ? 0 : 16);
  unsigned short* lds = pbuf[wave];

  for (int it = 0; it < 98; ++it) {
    const int n0 = it * 32;
    // K fragments for two 16-column score tiles
    Frag16 bk0, bk1;
    const unsigned short* kp0 = kbase + (size_t)(n0 + l15) * 128;
    const unsigned short* kp1 = kbase + (size_t)(n0 + 16 + l15) * 128;
    bk0.q[0] = *(const uint4*)kp0; bk0.q[1] = *(const uint4*)(kp0 + 8);
    bk1.q[0] = *(const uint4*)kp1; bk1.q[1] = *(const uint4*)(kp1 + 8);
    if (it + 1 < 98)
      __builtin_prefetch(kbase + (size_t)(n0 + 32 + l15) * 128, 0, 1);

    v8f z = (v8f)(0.f);
    v8f s0 = __builtin_amdgcn_wmma_f32_16x16x32_bf16(false, aq.v, false, bk0.v, (short)0, z, false, false);
    v8f s1 = __builtin_amdgcn_wmma_f32_16x16x32_bf16(false, aq.v, false, bk1.v, (short)0, z, false, false);

    // Issue V-fragment loads NOW: the softmax VALU block below hides their latency.
    Frag16 bv0, bv1;
    const unsigned short* vp0 = vbase + (size_t)l15 * HW + n0;
    const unsigned short* vp1 = vbase + (size_t)(16 + l15) * HW + n0;
    bv0.q[0] = *(const uint4*)vp0; bv0.q[1] = *(const uint4*)(vp0 + 8);
    bv1.q[0] = *(const uint4*)vp1; bv1.q[1] = *(const uint4*)(vp1 + 8);

    // online softmax: rows live in vgpr index r, columns in lanes (16-lane halves)
#pragma unroll
    for (int r = 0; r < 8; ++r) {
      float v0 = s0[r] * scale, v1 = s1[r] * scale;
      float tmax = fmaxf(v0, v1);
      tmax = fmaxf(tmax, __shfl_xor(tmax, 1, 32));
      tmax = fmaxf(tmax, __shfl_xor(tmax, 2, 32));
      tmax = fmaxf(tmax, __shfl_xor(tmax, 4, 32));
      tmax = fmaxf(tmax, __shfl_xor(tmax, 8, 32));
      float mn = fmaxf(mrow[r], tmax);
      float p0 = __expf(v0 - mn);
      float p1 = __expf(v1 - mn);
      float ps = p0 + p1;
      ps += __shfl_xor(ps, 1, 32);
      ps += __shfl_xor(ps, 2, 32);
      ps += __shfl_xor(ps, 4, 32);
      ps += __shfl_xor(ps, 8, 32);
      float corr = __expf(mrow[r] - mn);
      lrow[r] = lrow[r] * corr + ps;
      mrow[r] = mn;
      acc0[r] *= corr;
      acc1[r] *= corr;
      const int m = r + (lo ? 0 : 8);
      lds[m * 32 + l15] = f2bf(p0);            // D-layout -> row-major P tile in LDS
      lds[m * 32 + 16 + l15] = f2bf(p1);
    }
    __syncthreads();
    // reload P as A-fragment (16x32)
    Frag16 ap;
    {
      const unsigned short* lp = lds + l15 * 32 + (lo ? 0 : 16);
      ap.q[0] = *(const uint4*)lp;
      ap.q[1] = *(const uint4*)(lp + 8);
    }
    __syncthreads();

    acc0 = __builtin_amdgcn_wmma_f32_16x16x32_bf16(false, ap.v, false, bv0.v, (short)0, acc0, false, false);
    acc1 = __builtin_amdgcn_wmma_f32_16x16x32_bf16(false, ap.v, false, bv1.v, (short)0, acc1, false, false);
  }

#pragma unroll
  for (int r = 0; r < 8; ++r) {
    float inv = 1.0f / lrow[r];
    const int m = m0 + r + (lo ? 0 : 8);
    unsigned short* op = outT + ((size_t)(b * HW + m)) * 128 + h * 32 + l15;
    op[0] = f2bf(acc0[r] * inv);
    op[16] = f2bf(acc1[r] * inv);
  }
}

// ---------------- host side ----------------
extern "C" void kernel_launch(void* const* d_in, const int* in_sizes, int n_in,
                              void* d_out, int out_size, void* d_ws, size_t ws_size,
                              hipStream_t stream) {
  (void)in_sizes; (void)n_in; (void)out_size; (void)ws_size;
  const float* x     = (const float*)d_in[0];
  const float* dw_w  = (const float*)d_in[1];
  const float* dw_b  = (const float*)d_in[2];
  const float* ln_g  = (const float*)d_in[3];
  const float* ln_b  = (const float*)d_in[4];
  const float* off_w = (const float*)d_in[5];
  const float* q_w   = (const float*)d_in[6];
  const float* q_b   = (const float*)d_in[7];
  const float* k_w   = (const float*)d_in[8];
  const float* k_b   = (const float*)d_in[9];
  const float* v_w   = (const float*)d_in[10];
  const float* v_b   = (const float*)d_in[11];
  const float* o_w   = (const float*)d_in[12];
  const float* o_b   = (const float*)d_in[13];

  char* p = (char*)d_ws;
  auto carve = [&](size_t bytes) -> void* {
    void* r = (void*)p;
    p += (bytes + 255) & ~(size_t)255;
    return r;
  };
  unsigned short* wAll = (unsigned short*)carve((size_t)4 * 16384 * 2);
  unsigned short* xT   = (unsigned short*)carve((size_t)BB * HW * 128 * 2);
  float*          qf   = (float*)carve((size_t)BB * 128 * HW * 4);
  unsigned short* qT   = (unsigned short*)carve((size_t)BB * HW * 128 * 2);
  float*          hn   = (float*)carve((size_t)8 * HW * 64 * 4);
  float*          pos  = (float*)carve((size_t)8 * HW * 2 * 4);
  unsigned short* xsT  = (unsigned short*)carve((size_t)BB * HW * 128 * 2);
  unsigned short* kTb  = (unsigned short*)carve((size_t)BB * HW * 128 * 2);
  unsigned short* vb   = (unsigned short*)carve((size_t)BB * 128 * HW * 2);
  unsigned short* oT   = (unsigned short*)carve((size_t)BB * HW * 128 * 2);

  unsigned short* wq = wAll;
  unsigned short* wk = wAll + 16384;
  unsigned short* wv = wAll + 2 * 16384;
  unsigned short* wo = wAll + 3 * 16384;

  cvt_w_kernel<<<256, 256, 0, stream>>>(q_w, k_w, v_w, o_w, wAll);
  xpose_kernel<<<(BB * HW * 128 + 255) / 256, 256, 0, stream>>>(x, xT);

  // q projection: fp32 (offset branch) + transposed bf16 (attention A-operand)
  gemm128_bf16<<<dim3(49, 4), 128, 0, stream>>>(wq, xT, q_b, HW, qf, nullptr, qT);

  // offset branch + deformable gather
  dwconv7_kernel<<<8 * HW, 64, 0, stream>>>(qf, dw_w, dw_b, hn);
  lnoff_kernel<<<8 * HW, 64, 0, stream>>>(hn, ln_g, ln_b, off_w, pos);
  gather_kernel<<<(8 * HW + 255) / 256, 256, 0, stream>>>(x, pos, xsT);

  // k projection -> kT [B][n][128]; v projection -> v [B][128][n]
  gemm128_bf16<<<dim3(49, 4), 128, 0, stream>>>(wk, xsT, k_b, HW, nullptr, nullptr, kTb);
  gemm128_bf16<<<dim3(49, 4), 128, 0, stream>>>(wv, xsT, v_b, HW, nullptr, vb, nullptr);

  // flash attention -> outT [B][HW][128]
  attn_kernel<<<784, 128, 0, stream>>>(qT, kTb, vb, oT);

  // output projection -> d_out fp32 [B][128][H][W]
  gemm128_bf16<<<dim3(49, 4), 128, 0, stream>>>(wo, oT, o_b, HW, (float*)d_out, nullptr, nullptr);
}
